// GraphSAGEClassifier_67216238182899
// MI455X (gfx1250) — matrix operand order repaired and checked by reading, not compiled, not run on hardware
//
#include <hip/hip_runtime.h>
#include <hip/hip_bf16.h>

// GraphSAGE 2-layer + BN + ReLU + head for gfx1250 (MI455X).
// Scatter/gather is the bandwidth-bound part (E*128*4B ~ 410MB/layer vs 23.3TB/s);
// GEMMs (~5 GFLOP total) run on v_wmma_f32_16x16x32_bf16 (fp32 accumulate).

typedef __attribute__((ext_vector_type(16))) __bf16 v16bf;
typedef __attribute__((ext_vector_type(8)))  float  v8f;

#define NN   50000
#define NE   800000
#define DIM  128
#define HID2 64
#define NCLS 10

__device__ __forceinline__ int clampi(int v, int lo, int hi) {
    return v < lo ? lo : (v > hi ? hi : v);
}

// ---------------------------------------------------------------------------
// In-degree count: cnt[d] += 1 per edge.
__global__ __launch_bounds__(256) void count_edges(const int* __restrict__ dst,
                                                   float* __restrict__ cnt) {
    int e = blockIdx.x * blockDim.x + threadIdx.x;
    if (e >= NE) return;
    int d = clampi(dst[e], 0, NN - 1);
    atomicAdd(&cnt[d], 1.0f);
}

// ---------------------------------------------------------------------------
// Feature scatter: agg[dst] += feat[src].  32 lanes/edge, float4 per lane.
// Accumulator buffer (25.6MB) is L2-resident -> atomics resolve in L2.
__global__ __launch_bounds__(256) void scatter_feats(const float* __restrict__ feat,
                                                     const int* __restrict__ src,
                                                     const int* __restrict__ dst,
                                                     float* __restrict__ agg) {
    long long t = (long long)blockIdx.x * blockDim.x + threadIdx.x;
    if (t >= (long long)NE * 32) return;
    int e = (int)(t >> 5);
    int c = ((int)t & 31) << 2;                 // feature offset 0..124
    int s = clampi(src[e], 0, NN - 1);
    int d = clampi(dst[e], 0, NN - 1);
    const float4 v = *reinterpret_cast<const float4*>(feat + (size_t)s * DIM + c);
    float* ap = agg + (size_t)d * DIM + c;
    atomicAdd(ap + 0, v.x);
    atomicAdd(ap + 1, v.y);
    atomicAdd(ap + 2, v.z);
    atomicAdd(ap + 3, v.w);
}

// ---------------------------------------------------------------------------
// Fused SAGE layer: hout = relu(bn( (agg/max(cnt,1)) @ wl^T + bl + self @ wr^T ))
// Block: 16-row M tile, OUT*2 threads = OUT/16 waves, each wave one 16-col tile.
// K = DIM = 128 -> 4 WMMA k-steps per operand matrix (bf16 16x16x32).
template <int OUT>
__global__ __launch_bounds__(OUT * 2) void sage_layer(
    const float* __restrict__ selff, const float* __restrict__ agg,
    const float* __restrict__ cnt,
    const float* __restrict__ wl, const float* __restrict__ bl,
    const float* __restrict__ wr,
    const float* __restrict__ g, const float* __restrict__ be,
    const float* __restrict__ mu, const float* __restrict__ va,
    float* __restrict__ hout) {
    __shared__ __bf16 sMean[16][DIM + 4];   // +4 bf16 pad: kill LDS bank conflicts
    __shared__ __bf16 sSelf[16][DIM + 4];

    const int row0 = blockIdx.x * 16;

    // Stage mean-aggregated tile and self tile into LDS as bf16.
    for (int i = threadIdx.x; i < 16 * DIM; i += OUT * 2) {
        int m = i >> 7, k = i & (DIM - 1);
        int r = row0 + m;
        float xv = 0.0f, mv = 0.0f;
        if (r < NN) {
            xv = selff[(size_t)r * DIM + k];
            mv = agg[(size_t)r * DIM + k] / fmaxf(cnt[r], 1.0f);
        }
        sSelf[m][k] = (__bf16)xv;
        sMean[m][k] = (__bf16)mv;
    }
    __syncthreads();

    const int wave = threadIdx.x >> 5;
    const int lane = threadIdx.x & 31;
    const int n0   = wave * 16;
    const int l16  = lane & 15;      // A: row M; B: col N; (different fragments)
    const int hi   = lane >> 4;      // lane-half selector

    v8f acc = {};
    const float* wlrow = wl + (size_t)(n0 + l16) * DIM;   // B[k][n] = w[n][k]
    const float* wrrow = wr + (size_t)(n0 + l16) * DIM;

    // mean @ wl^T
    #pragma unroll
    for (int kk = 0; kk < DIM; kk += 32) {
        v16bf a, b;
        // A-fragment: lane-half hi holds K = hi*16 + 0..15 contiguous (matches B).
        #pragma unroll
        for (int j = 0; j < 8; ++j) {
            int kb = kk + hi * 16 + 2 * j;
            a[2 * j]     = sMean[l16][kb];
            a[2 * j + 1] = sMean[l16][kb + 1];
        }
        // B-fragment (32x16): lanes 0-15 hold K=kk..kk+15, lanes 16-31 K=kk+16..kk+31,
        // contiguous in K for a fixed column -> 4x float4 from the weight row.
        const float4* wp = reinterpret_cast<const float4*>(wlrow + kk + hi * 16);
        #pragma unroll
        for (int j = 0; j < 4; ++j) {
            float4 q = wp[j];
            b[4 * j + 0] = (__bf16)q.x;
            b[4 * j + 1] = (__bf16)q.y;
            b[4 * j + 2] = (__bf16)q.z;
            b[4 * j + 3] = (__bf16)q.w;
        }
        acc = __builtin_amdgcn_wmma_f32_16x16x32_bf16(false, a, false, b,
                                                      (short)0, acc, false, false);
    }

    // self @ wr^T (accumulates into the same C)
    #pragma unroll
    for (int kk = 0; kk < DIM; kk += 32) {
        v16bf a, b;
        #pragma unroll
        for (int j = 0; j < 8; ++j) {
            int kb = kk + hi * 16 + 2 * j;
            a[2 * j]     = sSelf[l16][kb];
            a[2 * j + 1] = sSelf[l16][kb + 1];
        }
        const float4* wp = reinterpret_cast<const float4*>(wrrow + kk + hi * 16);
        #pragma unroll
        for (int j = 0; j < 4; ++j) {
            float4 q = wp[j];
            b[4 * j + 0] = (__bf16)q.x;
            b[4 * j + 1] = (__bf16)q.y;
            b[4 * j + 2] = (__bf16)q.z;
            b[4 * j + 3] = (__bf16)q.w;
        }
        acc = __builtin_amdgcn_wmma_f32_16x16x32_bf16(false, a, false, b,
                                                      (short)0, acc, false, false);
    }

    // Epilogue: bias + BN(eval) + ReLU.  C layout: n = lane&15, m = r + 8*hi.
    const int n = n0 + l16;
    const float scale = g[n] * rsqrtf(va[n] + 1e-5f);
    const float bias  = bl[n];
    const float mm    = mu[n];
    const float bb    = be[n];
    #pragma unroll
    for (int r = 0; r < 8; ++r) {
        int m  = r + 8 * hi;
        int rr = row0 + m;
        if (rr < NN) {
            float v = acc[r] + bias;
            v = (v - mm) * scale + bb;
            hout[(size_t)rr * OUT + n] = fmaxf(v, 0.0f);
        }
    }
}

// ---------------------------------------------------------------------------
// Head: out[n, c] = h2[n, :] . head_w[c, :] + head_b[c]   (N x 10, K = 64)
__global__ __launch_bounds__(256) void head_kernel(const float* __restrict__ h2,
                                                   const float* __restrict__ hw,
                                                   const float* __restrict__ hb,
                                                   float* __restrict__ out) {
    int t = blockIdx.x * blockDim.x + threadIdx.x;
    if (t >= NN * NCLS) return;
    int node = t / NCLS, cls = t % NCLS;
    const float4* hp = reinterpret_cast<const float4*>(h2 + (size_t)node * HID2);
    const float4* wp = reinterpret_cast<const float4*>(hw + (size_t)cls * HID2);
    float acc = 0.0f;
    #pragma unroll
    for (int j = 0; j < HID2 / 4; ++j) {
        float4 a = hp[j], b = wp[j];
        acc += a.x * b.x + a.y * b.y + a.z * b.z + a.w * b.w;
    }
    out[t] = acc + hb[cls];
}

// ---------------------------------------------------------------------------
static inline size_t align256(size_t v) { return (v + 255) & ~(size_t)255; }

extern "C" void kernel_launch(void* const* d_in, const int* in_sizes, int n_in,
                              void* d_out, int out_size, void* d_ws, size_t ws_size,
                              hipStream_t stream) {
    const float* x   = (const float*)d_in[0];
    const int*   ei  = (const int*)d_in[1];
    const float* w1l = (const float*)d_in[2];
    const float* b1l = (const float*)d_in[3];
    const float* w1r = (const float*)d_in[4];
    const float* g1  = (const float*)d_in[5];
    const float* be1 = (const float*)d_in[6];
    const float* mu1 = (const float*)d_in[7];
    const float* va1 = (const float*)d_in[8];
    const float* w2l = (const float*)d_in[9];
    const float* b2l = (const float*)d_in[10];
    const float* w2r = (const float*)d_in[11];
    const float* g2  = (const float*)d_in[12];
    const float* be2 = (const float*)d_in[13];
    const float* mu2 = (const float*)d_in[14];
    const float* va2 = (const float*)d_in[15];
    const float* hw  = (const float*)d_in[16];
    const float* hb  = (const float*)d_in[17];
    float* out = (float*)d_out;

    const int* src = ei;        // edge_index row 0
    const int* dst = ei + NE;   // edge_index row 1

    // Workspace carve-up (~64 MB total).
    char* ws = (char*)d_ws;
    size_t off = 0;
    float* cnt = (float*)(ws + off); off += align256((size_t)NN * sizeof(float));
    float* agg = (float*)(ws + off); off += align256((size_t)NN * DIM * sizeof(float));
    float* h1  = (float*)(ws + off); off += align256((size_t)NN * DIM * sizeof(float));
    float* h2b = (float*)(ws + off); off += align256((size_t)NN * HID2 * sizeof(float));
    (void)ws_size; (void)n_in; (void)in_sizes; (void)out_size;

    const long long scatter_threads = (long long)NE * 32;
    const int scatter_blocks = (int)((scatter_threads + 255) / 256);
    const int mtiles = (NN + 15) / 16;

    // ---- Layer 1 ----
    hipMemsetAsync(cnt, 0, (size_t)NN * sizeof(float), stream);
    hipMemsetAsync(agg, 0, (size_t)NN * DIM * sizeof(float), stream);
    count_edges<<<(NE + 255) / 256, 256, 0, stream>>>(dst, cnt);
    scatter_feats<<<scatter_blocks, 256, 0, stream>>>(x, src, dst, agg);
    sage_layer<DIM><<<mtiles, DIM * 2, 0, stream>>>(
        x, agg, cnt, w1l, b1l, w1r, g1, be1, mu1, va1, h1);

    // ---- Layer 2 ----
    hipMemsetAsync(agg, 0, (size_t)NN * DIM * sizeof(float), stream);
    scatter_feats<<<scatter_blocks, 256, 0, stream>>>(h1, src, dst, agg);
    sage_layer<HID2><<<mtiles, HID2 * 2, 0, stream>>>(
        h1, agg, cnt, w2l, b2l, w2r, g2, be2, mu2, va2, h2b);

    // ---- Head ----
    head_kernel<<<(NN * NCLS + 255) / 256, 256, 0, stream>>>(h2b, hw, hb, out);
}
